// SpikeDecoder_35150012351092
// MI455X (gfx1250) — matrix operand order repaired and checked by reading, not compiled, not run on hardware
//
#include <hip/hip_runtime.h>
#include <hip/hip_bf16.h>

#define T_STEPS 512
#define BATCH   256
#define N1      512
#define N2      256
#define N3      64
#define N4      784
#define BETA_LIF 0.9f
#define BETA_LI  0.95f
#define LN_EPS   1e-5f

typedef __attribute__((ext_vector_type(16))) _Float16 v16h;
typedef __attribute__((ext_vector_type(8)))  _Float16 v8h;
typedef __attribute__((ext_vector_type(8)))  float    v8f;

union V16 { v16h v; v8h h[2]; };

#define A_PITCH 264  // 256 + 8 halfs pad -> conflict-free ds_load_b128

// ---------------------------------------------------------------------------
// Kernel 0: convert W_lif [N1,N2] and W_li [N2,N1] f32 -> f16 (row-major kept)
// ---------------------------------------------------------------------------
__global__ void prep_weights(const float* __restrict__ Wlif,
                             const float* __restrict__ Wli,
                             _Float16* __restrict__ WlifH,
                             _Float16* __restrict__ WliH) {
    int i = blockIdx.x * blockDim.x + threadIdx.x;   // 0 .. 131071
    WlifH[i] = (_Float16)Wlif[i];
    WliH[i]  = (_Float16)Wli[i];
}

// ---------------------------------------------------------------------------
// Kernel 1: fused LIF scan. Grid = BATCH/16 blocks of 512 threads (16 waves).
// Wave w owns output cols [32w, 32w+32) (two 16x16 WMMA tiles) for this
// block's 16 batch rows. Membrane + weighted spike sum stay in VGPRs for all
// T steps; per-step spike slice double-buffered in LDS as f16.
// ---------------------------------------------------------------------------
__global__ __launch_bounds__(512, 1) void lif_scan(
    const float*    __restrict__ spk,    // [T,B,N2] f32 (0/1)
    const _Float16* __restrict__ WlifH,  // [N1,N2] f16
    const float*    __restrict__ b_lif,  // [N1]
    _Float16*       __restrict__ S_h)    // [B,N1] f16 out: sum_t c_t * spike
{
    __shared__ _Float16 aBuf[2][16 * A_PITCH];

    const int tid    = threadIdx.x;
    const int wave   = tid >> 5;
    const int lane   = tid & 31;
    const int laneLo = lane & 15;
    const int hi     = lane >> 4;
    const int b0     = blockIdx.x * 16;
    const int nb0    = wave * 32;

    // ---- hoist W fragments (B operand in this code's contiguous k order: lanes0-15 K=0..15, lanes16-31
    //      K=16..31, N = lane&15) for all 8 K-steps of both tiles -----------
    v16h wfrag[2][8];
#pragma unroll
    for (int j = 0; j < 2; ++j) {
        const int n = nb0 + j * 16 + laneLo;
#pragma unroll
        for (int k = 0; k < 8; ++k) {
            const _Float16* p = WlifH + (size_t)n * N2 + k * 32 + hi * 16;
            wfrag[j][k] = *(const v16h*)p;           // 32B contiguous per lane
        }
    }
    float bias[2];
    bias[0] = b_lif[nb0 + laneLo];
    bias[1] = b_lif[nb0 + 16 + laneLo];

    float mem[2][8], sacc[2][8];
#pragma unroll
    for (int j = 0; j < 2; ++j)
#pragma unroll
        for (int i = 0; i < 8; ++i) { mem[j][i] = 0.0f; sacc[j][i] = 0.0f; }

    // ---- prefetch lane assignment: wave w loads row w, lane loads 8 floats
    const int prow = wave;          // 0..15
    const int pcol = lane * 8;      // 0..248

    // prologue: stage t = 0
    {
        const float* s = spk + (size_t)(b0 + prow) * N2 + pcol;
        float4 f0 = *(const float4*)s;
        float4 f1 = *(const float4*)(s + 4);
        v8h h;
        h[0] = (_Float16)f0.x; h[1] = (_Float16)f0.y;
        h[2] = (_Float16)f0.z; h[3] = (_Float16)f0.w;
        h[4] = (_Float16)f1.x; h[5] = (_Float16)f1.y;
        h[6] = (_Float16)f1.z; h[7] = (_Float16)f1.w;
        *(v8h*)&aBuf[0][prow * A_PITCH + pcol] = h;
    }
    __syncthreads();

    const float invBetaLi = 1.0f / BETA_LI;
    const float kcoef = 1.0f / ((1.0f - BETA_LI) * (float)T_STEPS);
    float p = __powf(BETA_LI, (float)T_STEPS);   // p_t = beta_li^(T-t)

    for (int t = 0; t < T_STEPS; ++t) {
        // prefetch t+1 slice into registers (overlaps WMMA below)
        float4 f0 = {0, 0, 0, 0}, f1 = {0, 0, 0, 0};
        const bool pf = (t + 1) < T_STEPS;
        if (pf) {
            const float* s = spk + (size_t)(t + 1) * BATCH * N2
                               + (size_t)(b0 + prow) * N2 + pcol;
            f0 = *(const float4*)s;
            f1 = *(const float4*)(s + 4);
        }

        // ---- cur = spike_slice x W_lif^T + bias (accumulated over K=256) --
        v8f cur0, cur1;
#pragma unroll
        for (int i = 0; i < 8; ++i) { cur0[i] = bias[0]; cur1[i] = bias[1]; }

        const _Float16* ab = &aBuf[t & 1][0];
#pragma unroll
        for (int k = 0; k < 8; ++k) {
            // A operand, same contiguous k order as B: lanes0-15 M=lane, K = 0..15; lanes16-31
            // M=lane-16, K = 16..31
            const _Float16* ap = ab + laneLo * A_PITCH + k * 32 + hi * 16;
            V16 a;
            a.h[0] = *(const v8h*)ap;
            a.h[1] = *(const v8h*)(ap + 8);
            cur0 = __builtin_amdgcn_wmma_f32_16x16x32_f16(
                false, a.v, false, wfrag[0][k], (short)0, cur0, false, false);
            cur1 = __builtin_amdgcn_wmma_f32_16x16x32_f16(
                false, a.v, false, wfrag[1][k], (short)0, cur1, false, false);
        }

        // ---- LIF update in accumulator layout ----------------------------
        const float c_t = (1.0f - p) * kcoef;
#pragma unroll
        for (int i = 0; i < 8; ++i) {
            float m0 = mem[0][i], m1 = mem[1][i];
            const float r0 = (m0 > 1.0f) ? 1.0f : 0.0f;   // reset = prev spike
            const float r1 = (m1 > 1.0f) ? 1.0f : 0.0f;
            m0 = BETA_LIF * m0 + cur0[i] - r0;
            m1 = BETA_LIF * m1 + cur1[i] - r1;
            mem[0][i] = m0; mem[1][i] = m1;
            if (m0 > 1.0f) sacc[0][i] += c_t;             // S += c_t * spike
            if (m1 > 1.0f) sacc[1][i] += c_t;
        }
        p *= invBetaLi;

        // ---- stage t+1 into the other LDS buffer -------------------------
        if (pf) {
            v8h h;
            h[0] = (_Float16)f0.x; h[1] = (_Float16)f0.y;
            h[2] = (_Float16)f0.z; h[3] = (_Float16)f0.w;
            h[4] = (_Float16)f1.x; h[5] = (_Float16)f1.y;
            h[6] = (_Float16)f1.z; h[7] = (_Float16)f1.w;
            *(v8h*)&aBuf[(t + 1) & 1][prow * A_PITCH + pcol] = h;
        }
        __syncthreads();
    }

    // ---- write weighted spike sum S (D-layout: vgpr v -> M = v + hi*8) ---
#pragma unroll
    for (int j = 0; j < 2; ++j)
#pragma unroll
        for (int v = 0; v < 8; ++v) {
            const int row = b0 + v + hi * 8;
            const int col = nb0 + j * 16 + laneLo;
            S_h[(size_t)row * N1 + col] = (_Float16)sacc[j][v];
        }
}

// ---------------------------------------------------------------------------
// Kernel 2: x[B,N2] = S[B,N1] @ W_li^T + (sum_t c_t) * b_li   (WMMA f16)
// 256 output tiles of 16x16; 8 waves/block, 32 blocks.
// ---------------------------------------------------------------------------
__global__ __launch_bounds__(256, 1) void li_gemm(
    const _Float16* __restrict__ S_h,   // [B,N1]
    const _Float16* __restrict__ WliH,  // [N2,N1]
    const float*    __restrict__ b_li,  // [N2]
    float*          __restrict__ x)     // [B,N2]
{
    const int wave   = threadIdx.x >> 5;
    const int lane   = threadIdx.x & 31;
    const int laneLo = lane & 15;
    const int hi     = lane >> 4;
    const int tile   = blockIdx.x * 8 + wave;   // 0..255
    const int tb     = (tile >> 4) * 16;        // batch-row base
    const int tn     = (tile & 15) * 16;        // out-col base

    v8f acc;
#pragma unroll
    for (int i = 0; i < 8; ++i) acc[i] = 0.0f;

#pragma unroll
    for (int k = 0; k < 16; ++k) {              // K = 512
        const _Float16* ap = S_h + (size_t)(tb + laneLo) * N1 + k * 32 + hi * 16;
        V16 a;
        a.h[0] = *(const v8h*)ap;
        a.h[1] = *(const v8h*)(ap + 8);
        const _Float16* bp = WliH + (size_t)(tn + laneLo) * N1 + k * 32 + hi * 16;
        const v16h b = *(const v16h*)bp;
        acc = __builtin_amdgcn_wmma_f32_16x16x32_f16(
            false, a.v, false, b, (short)0, acc, false, false);
    }

    // Cb = sum_t c_t = kcoef * (T - beta*(1-beta^T)/(1-beta))
    const float kcoef = 1.0f / ((1.0f - BETA_LI) * (float)T_STEPS);
    const float bT = __powf(BETA_LI, (float)T_STEPS);
    const float Cb = kcoef * ((float)T_STEPS
                              - BETA_LI * (1.0f - bT) / (1.0f - BETA_LI));

    const int col = tn + laneLo;
    const float bo = Cb * b_li[col];
#pragma unroll
    for (int v = 0; v < 8; ++v) {
        const int row = tb + v + hi * 8;
        x[(size_t)row * N2 + col] = acc[v] + bo;
    }
}

// ---------------------------------------------------------------------------
// Kernel 3: h = relu(x@W1^T + b1); y = h@W2^T + b2; LayerNorm over N4.
// One block (256 thr) per batch row. Tiny FLOPs -> plain f32 VALU (exact).
// ---------------------------------------------------------------------------
__global__ __launch_bounds__(256) void mlp_ln(
    const float* __restrict__ x,   // [B,N2]
    const float* __restrict__ W1, const float* __restrict__ b1,
    const float* __restrict__ W2, const float* __restrict__ b2,
    const float* __restrict__ g,  const float* __restrict__ be,
    float* __restrict__ out)       // [B,N4]
{
    __shared__ float xs[N2];
    __shared__ float hs[N3];
    __shared__ float red[256];

    const int b = blockIdx.x, tid = threadIdx.x;
    xs[tid] = x[(size_t)b * N2 + tid];
    __syncthreads();

    if (tid < N3) {
        float acc = b1[tid];
        for (int i = 0; i < N2; ++i) acc += xs[i] * W1[tid * N2 + i];
        hs[tid] = fmaxf(acc, 0.0f);
    }
    __syncthreads();

    float ys[4];
    float sum = 0.0f, sq = 0.0f;
#pragma unroll
    for (int r = 0; r < 4; ++r) {
        const int n = tid + r * 256;
        float v = 0.0f;
        if (n < N4) {
            v = b2[n];
            for (int i = 0; i < N3; ++i) v += hs[i] * W2[n * N3 + i];
            sum += v; sq += v * v;
        }
        ys[r] = v;
    }

    red[tid] = sum; __syncthreads();
    for (int s = 128; s > 0; s >>= 1) {
        if (tid < s) red[tid] += red[tid + s];
        __syncthreads();
    }
    const float mu = red[0] / (float)N4;
    __syncthreads();

    red[tid] = sq; __syncthreads();
    for (int s = 128; s > 0; s >>= 1) {
        if (tid < s) red[tid] += red[tid + s];
        __syncthreads();
    }
    const float var = red[0] / (float)N4 - mu * mu;
    const float inv = rsqrtf(var + LN_EPS);

#pragma unroll
    for (int r = 0; r < 4; ++r) {
        const int n = tid + r * 256;
        if (n < N4)
            out[(size_t)b * N4 + n] = (ys[r] - mu) * inv * g[n] + be[n];
    }
}

// ---------------------------------------------------------------------------
extern "C" void kernel_launch(void* const* d_in, const int* in_sizes, int n_in,
                              void* d_out, int out_size, void* d_ws, size_t ws_size,
                              hipStream_t stream) {
    const float* spk  = (const float*)d_in[0];   // [T,B,N2]
    const float* Wlif = (const float*)d_in[1];   // [N1,N2]
    const float* blif = (const float*)d_in[2];   // [N1]
    const float* Wli  = (const float*)d_in[3];   // [N2,N1]
    const float* bli  = (const float*)d_in[4];   // [N2]
    const float* W1   = (const float*)d_in[5];   // [N3,N2]
    const float* b1   = (const float*)d_in[6];   // [N3]
    const float* W2   = (const float*)d_in[7];   // [N4,N3]
    const float* b2   = (const float*)d_in[8];   // [N4]
    const float* lng  = (const float*)d_in[9];   // [N4]
    const float* lnb  = (const float*)d_in[10];  // [N4]
    float* out = (float*)d_out;

    char* ws = (char*)d_ws;
    _Float16* WlifH = (_Float16*)(ws);                    // 256 KB
    _Float16* WliH  = (_Float16*)(ws + (256u << 10));     // 256 KB
    _Float16* S_h   = (_Float16*)(ws + (512u << 10));     // 256 KB
    float*    xbuf  = (float*)   (ws + (768u << 10));     // 256 KB

    prep_weights<<<512, 256, 0, stream>>>(Wlif, Wli, WlifH, WliH);
    lif_scan<<<BATCH / 16, 512, 0, stream>>>(spk, WlifH, blif, S_h);
    li_gemm<<<32, 256, 0, stream>>>(S_h, WliH, bli, xbuf);
    mlp_ln<<<BATCH, 256, 0, stream>>>(xbuf, W1, b1, W2, b2, lng, lnb, out);
}
